// GQA_953482739765
// MI455X (gfx1250) — matrix operand order repaired and checked by reading, not compiled, not run on hardware
//
#include <hip/hip_runtime.h>
#include <stdint.h>

#define HIDDEN  2048
#define NHEADS  32
#define NGROUPS 8
#define HEADDIM 64
#define BATCH   2
#define SEQ     2048
#define MROWS   (BATCH * SEQ)        // 4096
#define KVDIM   (NGROUPS * HEADDIM)  // 512
#define PSTR    40                   // padded LDS row stride (halfwords): conflict-free, 16B aligned

#if defined(__has_builtin)
#if __has_builtin(__builtin_amdgcn_sched_barrier)
#define SCHED_BARRIER() __builtin_amdgcn_sched_barrier(0)
#endif
#endif
#ifndef SCHED_BARRIER
#define SCHED_BARRIER() do {} while (0)
#endif

typedef __attribute__((ext_vector_type(16))) __bf16 v16bf;
typedef __attribute__((ext_vector_type(8)))  float  v8f;

union Frag {
    v16bf v;
    uint4 q[2];
};

__device__ __forceinline__ uint16_t f2bf(float f) {
    uint32_t u = __float_as_uint(f);
    uint32_t r = u + 0x7FFFu + ((u >> 16) & 1u);  // round-to-nearest-even
    return (uint16_t)(r >> 16);
}

__device__ __forceinline__ v8f wmma_bf16(const Frag& a, const Frag& b, v8f c) {
    return __builtin_amdgcn_wmma_f32_16x16x32_bf16(
        /*neg_a=*/false, a.v, /*neg_b=*/false, b.v,
        /*c_mod=*/(short)0, c, /*reuse_a=*/false, /*reuse_b=*/false);
}

// A fragment (16x32, MxK): A row-major [*, ld] bf16; rows row0..row0+15, k in [kk, kk+32)
// ISA layout: lanes 0-15 -> K {kk..kk+7, kk+16..kk+23}; lanes 16-31 -> K {kk+8..kk+15, kk+24..kk+31}
__device__ __forceinline__ Frag load_a(const uint16_t* A, int ld, int row0, int kk,
                                       int lane16, int half) {
    Frag f;
    const uint16_t* p = A + (size_t)(row0 + lane16) * ld + kk + half * 8;
    f.q[0] = *(const uint4*)(p);
    f.q[1] = *(const uint4*)(p + 16);
    return f;
}

// B fragment (32x16, KxN) loaded from BT [N, ld] (row n holds K contiguous); cols n0..n0+15
// ISA layout: lanes 0-15 -> K {kk..kk+7, kk+16..kk+23}; lanes 16-31 -> K {kk+8..kk+15, kk+24..kk+31}
__device__ __forceinline__ Frag load_b(const uint16_t* BT, int ld, int n0, int kk,
                                       int lane16, int half) {
    Frag f;
    const uint16_t* p = BT + (size_t)(n0 + lane16) * ld + kk + half * 8;
    f.q[0] = *(const uint4*)(p);
    f.q[1] = *(const uint4*)(p + 16);
    return f;
}

// ---------------- elementwise converters ----------------

__global__ void f32_to_bf16_kernel(const float* __restrict__ in, uint16_t* __restrict__ out,
                                   size_t n) {
    size_t i = (size_t)blockIdx.x * blockDim.x + threadIdx.x;
    if (i < n) out[i] = f2bf(in[i]);
}

// W [K,N] f32 -> WT [N,K] bf16
__global__ void convT_kernel(const float* __restrict__ W, uint16_t* __restrict__ WT,
                             int K, int N) {
    size_t i = (size_t)blockIdx.x * blockDim.x + threadIdx.x;
    if (i >= (size_t)K * N) return;
    int k = (int)(i / N);
    int n = (int)(i % N);
    WT[(size_t)n * K + k] = f2bf(W[i]);
}

// V [(b*SEQ+s), KVDIM] bf16 -> VT [((b*NGROUPS+g)*HEADDIM+dd), SEQ] bf16
__global__ void transpose_v_kernel(const uint16_t* __restrict__ V, uint16_t* __restrict__ VT) {
    size_t i = (size_t)blockIdx.x * blockDim.x + threadIdx.x;
    if (i >= (size_t)MROWS * KVDIM) return;
    int s  = (int)(i % SEQ);
    int t  = (int)(i / SEQ);
    int dd = t % HEADDIM; t /= HEADDIM;
    int g  = t % NGROUPS;
    int b  = t / NGROUPS;
    VT[i] = V[(size_t)(b * SEQ + s) * KVDIM + g * HEADDIM + dd];
}

// ---------------- generic bf16 WMMA GEMM (ping-pong pipelined, K = HIDDEN) ----------------
// C[M,N] = A[M,K] * BT[N,K]^T + bias[N].
// Block = 128 threads = 4 waves stacked on M; each wave computes 32(M) x 64(N):
// 2 A-frags x 4 B-frags -> 8 WMMAs per 32-wide K step. Two register sets ping-pong
// with a peeled tail; a sched_barrier at the body end stops cross-iteration rotation
// that would split accumulator live ranges into v_mov chains.

__device__ __forceinline__ void g_load(Frag a[2], Frag b[4],
                                       const uint16_t* A, const uint16_t* BT,
                                       int mbase, int nbase, int kk,
                                       int lane16, int half) {
    a[0] = load_a(A, HIDDEN, mbase,      kk, lane16, half);
    a[1] = load_a(A, HIDDEN, mbase + 16, kk, lane16, half);
#pragma unroll
    for (int t = 0; t < 4; ++t) b[t] = load_b(BT, HIDDEN, nbase + t * 16, kk, lane16, half);
}

__device__ __forceinline__ void g_mma(v8f acc[2][4], const Frag a[2], const Frag b[4]) {
#pragma unroll
    for (int t = 0; t < 4; ++t) {
        acc[0][t] = wmma_bf16(a[0], b[t], acc[0][t]);
        acc[1][t] = wmma_bf16(a[1], b[t], acc[1][t]);
    }
}

__global__ __launch_bounds__(128)
void gemm_bf16_kernel(const uint16_t* __restrict__ A, const uint16_t* __restrict__ BT,
                      const float* __restrict__ bias,
                      uint16_t* __restrict__ outBf, float* __restrict__ outF,
                      int M, int N) {
    constexpr int K  = HIDDEN;
    const int lane   = threadIdx.x & 31;
    const int wave   = threadIdx.x >> 5;
    const int lane16 = lane & 15;
    const int half   = lane >> 4;
    const int mbase  = blockIdx.y * 128 + wave * 32;
    const int nbase  = blockIdx.x * 64;
    if (mbase >= M || nbase >= N) return;

    v8f acc[2][4];
#pragma unroll
    for (int w = 0; w < 2; ++w)
#pragma unroll
        for (int t = 0; t < 4; ++t)
            acc[w][t] = v8f{0.f,0.f,0.f,0.f,0.f,0.f,0.f,0.f};

    Frag aA[2], bA[4], aB[2], bB[4];
    g_load(aA, bA, A, BT, mbase, nbase, 0, lane16, half);

    int kk = 0;
    for (; kk < K - 64; kk += 64) {
        g_load(aB, bB, A, BT, mbase, nbase, kk + 32, lane16, half);
        // pull K-step kk+128 toward the caches (global_prefetch_b8)
        __builtin_prefetch(A  + (size_t)(mbase + lane16) * K + kk + 128, 0, 0);
        __builtin_prefetch(BT + (size_t)(nbase + lane16) * K + kk + 128, 0, 0);
        g_mma(acc, aA, bA);
        g_load(aA, bA, A, BT, mbase, nbase, kk + 64, lane16, half);
        g_mma(acc, aB, bB);
        SCHED_BARRIER();  // keep scheduling within the body: overlap is already explicit
    }
    // tail: kk == K-64
    g_load(aB, bB, A, BT, mbase, nbase, kk + 32, lane16, half);
    g_mma(acc, aA, bA);
    g_mma(acc, aB, bB);

#pragma unroll
    for (int w = 0; w < 2; ++w) {
#pragma unroll
        for (int t = 0; t < 4; ++t) {
            int col  = nbase + t * 16 + lane16;
            float bv = bias ? bias[col] : 0.f;
#pragma unroll
            for (int r = 0; r < 8; ++r) {
                int row = mbase + w * 16 + r + 8 * half;
                float v = acc[w][t][r] + bv;
                if (outBf) outBf[(size_t)row * N + col] = f2bf(v);
                if (outF)  outF [(size_t)row * N + col] = v;
            }
        }
    }
}

// ---------------- flash-style GQA attention (ping-pong on K-fragments) ----------------

__device__ __forceinline__ void attn_load_k(Frag bk[4], const uint16_t* Kb,
                                            int rowbase, int gcol, int lane16, int half) {
    bk[0] = load_b(Kb, KVDIM, rowbase,      gcol + 0,  lane16, half);
    bk[1] = load_b(Kb, KVDIM, rowbase,      gcol + 32, lane16, half);
    bk[2] = load_b(Kb, KVDIM, rowbase + 16, gcol + 0,  lane16, half);
    bk[3] = load_b(Kb, KVDIM, rowbase + 16, gcol + 32, lane16, half);
}

// process one 32-wide KV tile: scores -> online softmax -> LDS transpose -> P@V
__device__ __forceinline__ void attn_tile(const Frag& qf0, const Frag& qf1, const Frag bk[4],
                                          const uint16_t* __restrict__ VT, int vtrow, int kv0,
                                          v8f O[4], float rowm[8], float rowl[8],
                                          uint16_t* pt, int lane16, int half) {
    const float LOG2E = 1.44269504088896340736f;

    // V-fragments issued early: consumed only after the softmax VALU block
    Frag bv[4];
#pragma unroll
    for (int c = 0; c < 4; ++c) bv[c] = load_b(VT, SEQ, vtrow + c * 16, kv0, lane16, half);

    v8f s0 = v8f{0.f,0.f,0.f,0.f,0.f,0.f,0.f,0.f};
    v8f s1 = s0;
    s0 = wmma_bf16(qf0, bk[0], s0);
    s0 = wmma_bf16(qf1, bk[1], s0);
    s1 = wmma_bf16(qf0, bk[2], s1);
    s1 = wmma_bf16(qf1, bk[3], s1);

    // online softmax: rows live across the 16 lanes of each half-wave
#pragma unroll
    for (int r = 0; r < 8; ++r) {
        float a = s0[r] * 0.125f;   // 1/sqrt(64)
        float c = s1[r] * 0.125f;
        float t = fmaxf(a, c);
        t = fmaxf(t, __shfl_xor(t, 1, 16));
        t = fmaxf(t, __shfl_xor(t, 2, 16));
        t = fmaxf(t, __shfl_xor(t, 4, 16));
        t = fmaxf(t, __shfl_xor(t, 8, 16));
        float mnew = fmaxf(rowm[r], t);
        float corr = exp2f((rowm[r] - mnew) * LOG2E);
        float p0 = exp2f((a - mnew) * LOG2E);
        float p1 = exp2f((c - mnew) * LOG2E);
        float ps = p0 + p1;
        ps += __shfl_xor(ps, 1, 16);
        ps += __shfl_xor(ps, 2, 16);
        ps += __shfl_xor(ps, 4, 16);
        ps += __shfl_xor(ps, 8, 16);
        rowl[r] = rowl[r] * corr + ps;
        rowm[r] = mnew;
        O[0][r] *= corr; O[1][r] *= corr; O[2][r] *= corr; O[3][r] *= corr;
        int prow = r + 8 * half;
        pt[prow * PSTR + lane16]      = f2bf(p0);
        pt[prow * PSTR + 16 + lane16] = f2bf(p1);
    }

    __syncthreads();  // order LDS stores vs cross-lane A-frag reads (uniform everywhere)

    // re-read P (16x32) in A-fragment layout from LDS
    Frag pf;
    const uint16_t* pp = pt + lane16 * PSTR + half * 8;
    pf.q[0] = *(const uint4*)(pp);
    pf.q[1] = *(const uint4*)(pp + 16);

#pragma unroll
    for (int c = 0; c < 4; ++c) O[c] = wmma_bf16(pf, bv[c], O[c]);
}

// grid.x = b*NHEADS + h, grid.y = q-chunk of 128; 8 waves/block, 16 q-rows per wave.
// Writes attn output directly as A2[b*HIDDEN + h*64 + d][s]  (the swapaxes layout).
__global__ __launch_bounds__(256)
void attn_kernel(const uint16_t* __restrict__ Q, const uint16_t* __restrict__ Kb,
                 const uint16_t* __restrict__ VT, uint16_t* __restrict__ A2) {
    __shared__ __align__(16) uint16_t Pt[8][16 * PSTR];

    const int lane   = threadIdx.x & 31;
    const int wave   = threadIdx.x >> 5;
    const int lane16 = lane & 15;
    const int half   = lane >> 4;
    const int bh     = blockIdx.x;
    const int b      = bh >> 5;
    const int h      = bh & 31;
    const int g      = h >> 2;                       // rep = NHEADS/NGROUPS = 4
    const int qbase  = blockIdx.y * 128 + wave * 16; // q position within batch

    // Q tile 16x64 as two A-fragments, reused for every KV tile
    Frag qf0 = load_a(Q, HIDDEN, b * SEQ + qbase, h * HEADDIM + 0,  lane16, half);
    Frag qf1 = load_a(Q, HIDDEN, b * SEQ + qbase, h * HEADDIM + 32, lane16, half);

    v8f O[4];
#pragma unroll
    for (int c = 0; c < 4; ++c) O[c] = v8f{0.f,0.f,0.f,0.f,0.f,0.f,0.f,0.f};
    float rowm[8], rowl[8];
#pragma unroll
    for (int r = 0; r < 8; ++r) { rowm[r] = -__builtin_inff(); rowl[r] = 0.f; }

    uint16_t* pt = &Pt[wave][0];
    const int vtrow  = (b * NGROUPS + g) * HEADDIM;
    const int krow   = b * SEQ;
    const int gcol   = g * HEADDIM;

    // ping-pong the K-fragment sets so score WMMAs never wait on a just-issued load
    Frag bkA[4], bkB[4];
    attn_load_k(bkA, Kb, krow, gcol, lane16, half);

    int kv = 0;
    for (; kv < SEQ - 64; kv += 64) {
        attn_load_k(bkB, Kb, krow + kv + 32, gcol, lane16, half);
        attn_tile(qf0, qf1, bkA, VT, vtrow, kv,      O, rowm, rowl, pt, lane16, half);
        attn_load_k(bkA, Kb, krow + kv + 64, gcol, lane16, half);
        attn_tile(qf0, qf1, bkB, VT, vtrow, kv + 32, O, rowm, rowl, pt, lane16, half);
    }
    // tail: kv == SEQ-64
    attn_load_k(bkB, Kb, krow + kv + 32, gcol, lane16, half);
    attn_tile(qf0, qf1, bkA, VT, vtrow, kv,      O, rowm, rowl, pt, lane16, half);
    attn_tile(qf0, qf1, bkB, VT, vtrow, kv + 32, O, rowm, rowl, pt, lane16, half);

    // normalize and store in [b, h*64+d, s] layout (A operand for the final GEMM)
#pragma unroll
    for (int r = 0; r < 8; ++r) {
        float inv = 1.0f / rowl[r];
        int spos = qbase + r + 8 * half;
        size_t base = ((size_t)b * HIDDEN + h * HEADDIM) * SEQ + spos;
#pragma unroll
        for (int c = 0; c < 4; ++c)
            A2[base + (size_t)(c * 16 + lane16) * SEQ] = f2bf(O[c][r] * inv);
    }
}

// ---------------- host-side orchestration ----------------

extern "C" void kernel_launch(void* const* d_in, const int* in_sizes, int n_in,
                              void* d_out, int out_size, void* d_ws, size_t ws_size,
                              hipStream_t stream) {
    const float* X  = (const float*)d_in[0];
    const float* wq = (const float*)d_in[1];
    const float* bq = (const float*)d_in[2];
    const float* wk = (const float*)d_in[3];
    const float* bk = (const float*)d_in[4];
    const float* wv = (const float*)d_in[5];
    const float* bvv= (const float*)d_in[6];
    const float* wo = (const float*)d_in[7];
    const float* bo = (const float*)d_in[8];
    float* out = (float*)d_out;

    char* ws = (char*)d_ws;
    size_t off = 0;
    auto alloc = [&](size_t elems) {
        uint16_t* p = (uint16_t*)(ws + off);
        off += elems * sizeof(uint16_t);
        return p;
    };
    uint16_t* Xbf = alloc((size_t)MROWS  * HIDDEN);   // 16 MB
    uint16_t* Qbf = alloc((size_t)MROWS  * HIDDEN);   // 16 MB
    uint16_t* WqT = alloc((size_t)HIDDEN * HIDDEN);   //  8 MB
    uint16_t* WoT = alloc((size_t)HIDDEN * HIDDEN);   //  8 MB
    uint16_t* WkT = alloc((size_t)KVDIM  * HIDDEN);   //  2 MB
    uint16_t* WvT = alloc((size_t)KVDIM  * HIDDEN);   //  2 MB
    uint16_t* Kbf = alloc((size_t)MROWS  * KVDIM);    //  4 MB
    uint16_t* Vbf = alloc((size_t)MROWS  * KVDIM);    //  4 MB
    uint16_t* VTb = alloc((size_t)MROWS  * KVDIM);    //  4 MB
    uint16_t* A2  = alloc((size_t)MROWS  * HIDDEN);   // 16 MB  (total 80 MB)

    // bf16 conversions (+ weight transposes for contiguous B-fragment loads)
    {
        size_t n = (size_t)MROWS * HIDDEN;
        f32_to_bf16_kernel<<<(unsigned)((n + 255) / 256), 256, 0, stream>>>(X, Xbf, n);
    }
    convT_kernel<<<(unsigned)(((size_t)HIDDEN * HIDDEN + 255) / 256), 256, 0, stream>>>(wq, WqT, HIDDEN, HIDDEN);
    convT_kernel<<<(unsigned)(((size_t)HIDDEN * KVDIM  + 255) / 256), 256, 0, stream>>>(wk, WkT, HIDDEN, KVDIM);
    convT_kernel<<<(unsigned)(((size_t)HIDDEN * KVDIM  + 255) / 256), 256, 0, stream>>>(wv, WvT, HIDDEN, KVDIM);
    convT_kernel<<<(unsigned)(((size_t)HIDDEN * HIDDEN + 255) / 256), 256, 0, stream>>>(wo, WoT, HIDDEN, HIDDEN);

    dim3 blk(128);
    // Q/K/V projections (block tile 128M x 64N)
    gemm_bf16_kernel<<<dim3(HIDDEN / 64, MROWS / 128), blk, 0, stream>>>(
        Xbf, WqT, bq, Qbf, nullptr, MROWS, HIDDEN);
    gemm_bf16_kernel<<<dim3(KVDIM / 64, MROWS / 128), blk, 0, stream>>>(
        Xbf, WkT, bk, Kbf, nullptr, MROWS, KVDIM);
    gemm_bf16_kernel<<<dim3(KVDIM / 64, MROWS / 128), blk, 0, stream>>>(
        Xbf, WvT, bvv, Vbf, nullptr, MROWS, KVDIM);

    transpose_v_kernel<<<(unsigned)(((size_t)MROWS * KVDIM + 255) / 256), 256, 0, stream>>>(Vbf, VTb);

    // attention -> A2 in [b, h*64+d, s] layout
    attn_kernel<<<dim3(BATCH * NHEADS, SEQ / 128), 256, 0, stream>>>(Qbf, Kbf, VTb, A2);

    // final projection (contracts over s, faithful to reference's swapaxes+reshape)
    gemm_bf16_kernel<<<dim3(HIDDEN / 64, MROWS / 128), blk, 0, stream>>>(
        A2, WoT, bo, nullptr, out, MROWS, HIDDEN);
}
